// GCRNNGCN_66546223284412
// MI455X (gfx1250) — matrix operand order repaired and checked by reading, not compiled, not run on hardware
//
#include <hip/hip_runtime.h>
#include <cstdint>

typedef __attribute__((ext_vector_type(16))) __bf16 v16bf;
typedef __attribute__((ext_vector_type(8)))  float  v8f;

__device__ __forceinline__ unsigned short f2bf(float f) {
  unsigned u = __float_as_uint(f);
  u += 0x7FFFu + ((u >> 16) & 1u);           // round-to-nearest-even
  return (unsigned short)(u >> 16);
}
__device__ __forceinline__ float bf2f(unsigned short h) {
  return __uint_as_float(((unsigned)h) << 16);
}

union ABu { v16bf v; uint4 q[2]; unsigned short s[16]; };

// A operand: 16x32 bf16 tile from row-major [64][64], rows m0..m0+15, cols k0..k0+31
__device__ __forceinline__ v16bf loadA(const unsigned short* rm, int m0, int k0, int lane) {
  int m  = m0 + (lane & 15);
  int kh = (lane >> 4) << 3;                 // lanes 0-15: K 0..7 & 16..23 ; lanes 16-31: K 8..15 & 24..31
  const unsigned short* p = rm + m * 64 + k0 + kh;
  ABu u;
  u.q[0] = *(const uint4*)(p);
  u.q[1] = *(const uint4*)(p + 16);
  return u.v;
}
// B operand: 32x16 bf16 tile; storage is column-major mirror cm[n*64+k]
__device__ __forceinline__ v16bf loadB(const unsigned short* cm, int n0, int k0, int lane) {
  int n  = n0 + (lane & 15);
  int kb = k0 + ((lane >> 4) << 3);          // lanes 0-15: K 0..7 & 16..23 ; lanes 16-31: K 8..15 & 24..31
  const unsigned short* p = cm + n * 64 + kb;
  ABu u;
  u.q[0] = *(const uint4*)(p);
  u.q[1] = *(const uint4*)(p + 16);
  return u.v;
}
// one 16x16 output tile, K=64, A pair pre-hoisted in registers
__device__ __forceinline__ v8f tileMMr(v16bf a0, v16bf a1, const unsigned short* Bcm,
                                       int n0, int lane, v8f acc) {
  v16bf b0 = loadB(Bcm, n0, 0, lane);
  acc = __builtin_amdgcn_wmma_f32_16x16x32_bf16(false, a0, false, b0, (short)0, acc, false, false);
  v16bf b1 = loadB(Bcm, n0, 32, lane);
  acc = __builtin_amdgcn_wmma_f32_16x16x32_bf16(false, a1, false, b1, (short)0, acc, false, false);
  return acc;
}

#define SMEM_BYTES 122624

__global__ __launch_bounds__(128) void gcrnn_gcn_kernel(
    const float* __restrict__ x,  const float* __restrict__ ew,
    const float* __restrict__ wA, const float* __restrict__ bA,
    const float* __restrict__ wB, const float* __restrict__ bB,
    const float* __restrict__ w1, const float* __restrict__ b1,
    const float* __restrict__ w2, const float* __restrict__ b2,
    const float* __restrict__ wfc,const float* __restrict__ bfc,
    const long long* __restrict__ esrc, const long long* __restrict__ edst,
    float* __restrict__ out, int epg)
{
  extern __shared__ char smem[];
  unsigned short* Lrm  = (unsigned short*)(smem + 0);       // Lhat (later Agcn) bf16 row-major
  unsigned short* Wb   = (unsigned short*)(smem + 8192);    // wB[k] transposed, bf16 [j][i]
  unsigned short* Trm0 = (unsigned short*)(smem + 40960);   // Tx slots row-major bf16
  unsigned short* Trm1 = Trm0 + 4096;
  unsigned short* Trm2 = Trm0 + 8192;
  unsigned short* Tcm0 = (unsigned short*)(smem + 65536);   // Tx slots column-major bf16
  unsigned short* Tcm1 = Tcm0 + 4096;
  unsigned short* Tcm2 = Tcm0 + 8192;
  unsigned short* W1s  = (unsigned short*)(smem + 90112);   // w1 transposed [j][i]
  unsigned short* W2s  = (unsigned short*)(smem + 98304);   // w2 transposed [j][i], j<32
  float* Stage = (float*)(smem + 102400);                   // f32 staging (Lap build / Agcn / z2)
  float* Txa   = (float*)(smem + 118784);                   // scalar cheb [4][64]
  float* Deg   = (float*)(smem + 119808);
  float* Dinv  = (float*)(smem + 120064);
  float* Dinv2 = (float*)(smem + 120320);
  float* BAs   = (float*)(smem + 120576);
  float* BBs   = (float*)(smem + 120832);
  float* B1s   = (float*)(smem + 121088);
  float* B2s   = (float*)(smem + 121344);
  float* WAs   = (float*)(smem + 121472);                   // wA [k][j]
  float* Logit = (float*)(smem + 122496);

  const int tid  = threadIdx.x;
  const int lane = tid & 31;
  const int m0   = (tid >> 5) * 16;          // wave's row stripe
  const int g    = blockIdx.x;
  const size_t ebase = (size_t)g * (size_t)epg;
  const int nD  = (lane & 15);               // D/B tile: lane's N offset
  const int mbD = m0 + ((lane >> 4) << 3);   // D tile: lane's first M row

  // ---------------- Phase 0: stage weights, build dense scaled Laplacian ----------------
  for (int i = tid; i < 4096; i += 128) {
    Stage[i] = 0.0f;
    Trm0[i] = 0; Tcm0[i] = 0;                // h0 = 0
    int ii = i >> 6, jj = i & 63;
    W1s[jj * 64 + ii] = f2bf(w1[i]);
  }
  for (int i = tid; i < 16384; i += 128) {
    int k = i >> 12, rem = i & 4095, ii = rem >> 6, jj = rem & 63;
    Wb[k * 4096 + jj * 64 + ii] = f2bf(wB[i]);
  }
  for (int i = tid; i < 2048; i += 128) {
    int ii = i >> 5, jj = i & 31;
    W2s[jj * 64 + ii] = f2bf(w2[i]);
  }
  if (tid < 64) { BAs[tid]=bA[tid]; BBs[tid]=bB[tid]; B1s[tid]=b1[tid]; Deg[tid]=0.0f; }
  if (tid < 32) B2s[tid] = b2[tid];
  for (int i = tid; i < 256; i += 128) WAs[i] = wA[i];
  __syncthreads();
  for (int e = tid; e < epg; e += 128) {                 // deg = seg(ew, src)
    int s = (int)(esrc[ebase + e]) & 63;
    atomicAdd(&Deg[s], ew[ebase + e]);
  }
  __syncthreads();
  if (tid < 64) Dinv[tid] = (Deg[tid] > 0.0f) ? rsqrtf(Deg[tid]) : 0.0f;
  __syncthreads();
  for (int e = tid; e < epg; e += 128) {                 // Lhat[dst][src] += w_lap
    int s = (int)(esrc[ebase + e]) & 63;
    int d = (int)(edst[ebase + e]) & 63;
    float w = ew[ebase + e];
    atomicAdd(&Stage[d * 64 + s], -(Dinv[s] * w * Dinv[d]));
  }
  __syncthreads();
  for (int i = tid; i < 4096; i += 128) Lrm[i] = f2bf(Stage[i]);
  __syncthreads();

  // Lhat A-operand pair: invariant across the whole recurrence -> keep in registers
  v16bf aL0 = loadA(Lrm, m0, 0, lane);
  v16bf aL1 = loadA(Lrm, m0, 32, lane);

  // ---------------- Phase 1: GCRNN recurrence, 64 steps ----------------
  for (int t = 0; t < 64; ++t) {
    // scalar (1-channel) Chebyshev path on x_t
    if (tid < 64) Txa[tid] = x[((size_t)(g * 64 + tid)) * 64 + t];
    __syncthreads();
    if (tid < 64) {
      float s = 0.0f;
      for (int k = 0; k < 64; ++k) s += bf2f(Lrm[tid * 64 + k]) * Txa[k];
      Txa[64 + tid] = s;
    }
    __syncthreads();
    if (tid < 64) {
      float s = 0.0f;
      for (int k = 0; k < 64; ++k) s += bf2f(Lrm[tid * 64 + k]) * Txa[64 + k];
      Txa[128 + tid] = 2.0f * s - Txa[tid];
    }
    __syncthreads();
    if (tid < 64) {
      float s = 0.0f;
      for (int k = 0; k < 64; ++k) s += bf2f(Lrm[tid * 64 + k]) * Txa[128 + k];
      Txa[192 + tid] = 2.0f * s - Txa[64 + tid];
    }
    // Tx1 = Lhat @ h   (slot0 -> slot1)
    for (int i = 0; i < 4; ++i) {
      int n0 = i * 16, n = n0 + nD;
      v8f d = {};
      d = tileMMr(aL0, aL1, Tcm0, n0, lane, d);
      alignas(16) unsigned short tmp[8];
      for (int r = 0; r < 8; ++r) { unsigned short b = f2bf(d[r]); tmp[r] = b; Trm1[(mbD + r) * 64 + n] = b; }
      *(uint4*)(Tcm1 + n * 64 + mbD) = *(const uint4*)tmp;
    }
    __syncthreads();
    v8f acc[4] = {};                                     // bh accumulator, 16x64 stripe per wave
    {
      v16bf a0 = loadA(Trm0, m0, 0, lane), a1 = loadA(Trm0, m0, 32, lane);
      for (int i = 0; i < 4; ++i) acc[i] = tileMMr(a0, a1, Wb, i * 16, lane, acc[i]);        // Tx0 @ W0
    }
    {
      v16bf a0 = loadA(Trm1, m0, 0, lane), a1 = loadA(Trm1, m0, 32, lane);
      for (int i = 0; i < 4; ++i) acc[i] = tileMMr(a0, a1, Wb + 4096, i * 16, lane, acc[i]); // Tx1 @ W1
    }
    // Tx2 = 2*Lhat@Tx1 - Tx0  (fused in epilogue) -> slot2
    for (int i = 0; i < 4; ++i) {
      int n0 = i * 16, n = n0 + nD;
      v8f d = {};
      d = tileMMr(aL0, aL1, Tcm1, n0, lane, d);
      alignas(16) unsigned short tmp[8];
      for (int r = 0; r < 8; ++r) {
        float v = 2.0f * d[r] - bf2f(Tcm0[n * 64 + mbD + r]);
        unsigned short b = f2bf(v); tmp[r] = b; Trm2[(mbD + r) * 64 + n] = b;
      }
      *(uint4*)(Tcm2 + n * 64 + mbD) = *(const uint4*)tmp;
    }
    __syncthreads();
    {
      v16bf a0 = loadA(Trm2, m0, 0, lane), a1 = loadA(Trm2, m0, 32, lane);
      for (int i = 0; i < 4; ++i) acc[i] = tileMMr(a0, a1, Wb + 8192, i * 16, lane, acc[i]); // Tx2 @ W2
    }
    // Tx3 = 2*Lhat@Tx2 - Tx1 -> slot0 rm, own row stripe only (wave-private; no barrier)
    for (int i = 0; i < 4; ++i) {
      int n0 = i * 16, n = n0 + nD;
      v8f d = {};
      d = tileMMr(aL0, aL1, Tcm2, n0, lane, d);
      for (int r = 0; r < 8; ++r) {
        float v = 2.0f * d[r] - bf2f(Tcm1[n * 64 + mbD + r]);
        Trm0[(mbD + r) * 64 + n] = f2bf(v);
      }
    }
    {
      v16bf a0 = loadA(Trm0, m0, 0, lane), a1 = loadA(Trm0, m0, 32, lane);
      for (int i = 0; i < 4; ++i) acc[i] = tileMMr(a0, a1, Wb + 12288, i * 16, lane, acc[i]); // Tx3 @ W3
    }
    // h_new = sigmoid(a + bh); stores touch only this wave's stripe -> no barrier needed
    for (int i = 0; i < 4; ++i) {
      int n = i * 16 + nD;
      alignas(16) unsigned short tmp[8];
      for (int r = 0; r < 8; ++r) {
        int m = mbD + r;
        float av = BAs[n];
        for (int k = 0; k < 4; ++k) av += Txa[k * 64 + m] * WAs[k * 64 + n];
        float v = acc[i][r] + BBs[n] + av;
        float h = 1.0f / (1.0f + expf(-v));
        unsigned short b = f2bf(h); tmp[r] = b; Trm0[m * 64 + n] = b;
      }
      *(uint4*)(Tcm0 + n * 64 + mbD) = *(const uint4*)tmp;
    }
    __syncthreads();
  }

  // ---------------- Phase 2: two GCN layers + FC + log_softmax ----------------
  if (tid < 64) Deg[tid] = 0.0f;
  for (int i = tid; i < 4096; i += 128) Stage[i] = 0.0f;
  __syncthreads();
  for (int e = tid; e < epg; e += 128) {                 // deg = seg(ew, dst) + 1
    int d = (int)(edst[ebase + e]) & 63;
    atomicAdd(&Deg[d], ew[ebase + e]);
  }
  __syncthreads();
  if (tid < 64) { float di = rsqrtf(Deg[tid] + 1.0f); Dinv[tid] = di; Dinv2[tid] = di * di; }
  __syncthreads();
  for (int e = tid; e < epg; e += 128) {                 // Agcn[dst][src]
    int s = (int)(esrc[ebase + e]) & 63, d = (int)(edst[ebase + e]) & 63;
    atomicAdd(&Stage[d * 64 + s], Dinv[s] * ew[ebase + e] * Dinv[d]);
  }
  __syncthreads();
  for (int i = tid; i < 4096; i += 128) Lrm[i] = f2bf(Stage[i]);   // reuse Lrm for Agcn
  __syncthreads();
  v16bf aG0 = loadA(Lrm, m0, 0, lane), aG1 = loadA(Lrm, m0, 32, lane);  // Agcn A pair
  // xw = h @ w1 -> col-major
  {
    v16bf a0 = loadA(Trm0, m0, 0, lane), a1 = loadA(Trm0, m0, 32, lane);
    for (int i = 0; i < 4; ++i) {
      int n0 = i * 16, n = n0 + nD;
      v8f d = {};
      d = tileMMr(a0, a1, W1s, n0, lane, d);
      alignas(16) unsigned short tmp[8];
      for (int r = 0; r < 8; ++r) tmp[r] = f2bf(d[r]);
      *(uint4*)(Tcm1 + n * 64 + mbD) = *(const uint4*)tmp;
    }
  }
  __syncthreads();
  // z1 = relu(Agcn@xw + dinv2*xw + b1) -> row-major, own stripe
  for (int i = 0; i < 4; ++i) {
    int n0 = i * 16, n = n0 + nD;
    v8f d = {};
    d = tileMMr(aG0, aG1, Tcm1, n0, lane, d);
    for (int r = 0; r < 8; ++r) {
      int m = mbD + r;
      float v = d[r] + Dinv2[m] * bf2f(Tcm1[n * 64 + m]) + B1s[n];
      Trm1[m * 64 + n] = f2bf(fmaxf(v, 0.0f));
    }
  }
  // xw2 = z1 @ w2 (N=32): A is this wave's own z1 stripe -> no barrier needed
  {
    v16bf a0 = loadA(Trm1, m0, 0, lane), a1 = loadA(Trm1, m0, 32, lane);
    for (int i = 0; i < 2; ++i) {
      int n0 = i * 16, n = n0 + nD;
      v8f d = {};
      d = tileMMr(a0, a1, W2s, n0, lane, d);
      alignas(16) unsigned short tmp[8];
      for (int r = 0; r < 8; ++r) tmp[r] = f2bf(d[r]);
      *(uint4*)(Tcm2 + n * 64 + mbD) = *(const uint4*)tmp;
    }
  }
  __syncthreads();
  // z2 = relu(Agcn@xw2 + dinv2*xw2 + b2) -> Stage f32 [node][32]
  for (int i = 0; i < 2; ++i) {
    int n0 = i * 16, n = n0 + nD;
    v8f d = {};
    d = tileMMr(aG0, aG1, Tcm2, n0, lane, d);
    for (int r = 0; r < 8; ++r) {
      int m = mbD + r;
      float v = d[r] + Dinv2[m] * bf2f(Tcm2[n * 64 + m]) + B2s[n];
      Stage[m * 32 + n] = fmaxf(v, 0.0f);
    }
  }
  if (tid < 4) Logit[tid] = 0.0f;
  __syncthreads();
  // FC 2048x4 + log_softmax
  float p0 = 0, p1 = 0, p2 = 0, p3 = 0;
  for (int idx = tid; idx < 2048; idx += 128) {
    float v = Stage[idx];
    const float* wr = wfc + (size_t)idx * 4;
    p0 += v * wr[0]; p1 += v * wr[1]; p2 += v * wr[2]; p3 += v * wr[3];
  }
  atomicAdd(&Logit[0], p0); atomicAdd(&Logit[1], p1);
  atomicAdd(&Logit[2], p2); atomicAdd(&Logit[3], p3);
  __syncthreads();
  if (tid == 0) {
    float l[4]; float mx = -1e30f;
    for (int c = 0; c < 4; ++c) { l[c] = Logit[c] + bfc[c]; mx = fmaxf(mx, l[c]); }
    float se = 0.0f;
    for (int c = 0; c < 4; ++c) se += expf(l[c] - mx);
    float lse = logf(se);
    for (int c = 0; c < 4; ++c) out[(size_t)g * 4 + c] = l[c] - mx - lse;
  }
}

extern "C" void kernel_launch(void* const* d_in, const int* in_sizes, int n_in,
                              void* d_out, int out_size, void* d_ws, size_t ws_size,
                              hipStream_t stream) {
  const float* x   = (const float*)d_in[0];
  const float* ew  = (const float*)d_in[1];
  const float* wA  = (const float*)d_in[2];
  const float* bA  = (const float*)d_in[3];
  const float* wB  = (const float*)d_in[4];
  const float* bB  = (const float*)d_in[5];
  const float* w1  = (const float*)d_in[6];
  const float* b1  = (const float*)d_in[7];
  const float* w2  = (const float*)d_in[8];
  const float* b2  = (const float*)d_in[9];
  const float* wfc = (const float*)d_in[10];
  const float* bfc = (const float*)d_in[11];
  const long long* esrc = (const long long*)d_in[12];
  const long long* edst = (const long long*)d_in[13];

  int N   = in_sizes[0] / 64;   // T = 64
  int Bn  = N / 64;             // num_nodes = 64
  int epg = in_sizes[1] / Bn;   // edges per graph
  float* out = (float*)d_out;

  gcrnn_gcn_kernel<<<dim3(Bn), dim3(128), SMEM_BYTES, stream>>>(
      x, ew, wA, bA, wB, bB, w1, b1, w2, b2, wfc, bfc, esrc, edst, out, epg);
}
